// TransducerJoiner_57526791962920
// MI455X (gfx1250) — matrix operand order repaired and checked by reading, not compiled, not run on hardware
//
#include <hip/hip_runtime.h>

// Problem constants (match reference)
#define BB 8
#define TT 256
#define UU 64
#define DD 512
#define HH 1024

#define ROWS 32        // rows of the (t,u) tile per workgroup (half of U)
#define HCHUNK 256     // H processed per fused stage
#define SA_STRIDE 520  // 512 + 8 halves pad (stagger LDS banks, keeps 16B align)
#define SH_STRIDE 264  // 256 + 8 halves pad

typedef __attribute__((ext_vector_type(16))) _Float16 v16h;
typedef __attribute__((ext_vector_type(8)))  _Float16 v8h;
typedef __attribute__((ext_vector_type(8)))  float    v8f;

// Branch-free tanh. Prefer gfx1250's native v_tanh_f32 transcendental; fall
// back to exp2+rcp (still branchless, saturates correctly at +-inf).
__device__ __forceinline__ float fast_tanh(float x) {
#if __has_builtin(__builtin_amdgcn_tanhf)
  return __builtin_amdgcn_tanhf(x);
#else
  // tanh(x) = 1 - 2/(1 + 2^(2*log2(e)*x))
  float e = __builtin_amdgcn_exp2f(x * 2.88539008177792681472e0f); // 2*log2(e)
  return 1.0f - 2.0f * __builtin_amdgcn_rcpf(e + 1.0f);
#endif
}

// One-time: W1 (D,H) f32 -> w1t (H,D) f16 ; W2 (H,D) f32 -> w2t (D,H) f16.
// Transposed+converted so WMMA B-fragments are contiguous 32B per lane.
__global__ __launch_bounds__(512)
void joiner_prep(const float* __restrict__ W1, const float* __restrict__ W2,
                 _Float16* __restrict__ w1t, _Float16* __restrict__ w2t) {
  int i = blockIdx.x * blockDim.x + threadIdx.x;
  if (i >= DD * HH) return;
  int h = i / DD, d = i - h * DD;        // w1t[h][d] = W1[d][h]
  w1t[i] = (_Float16)W1[d * HH + h];
  int d2 = i / HH, h2 = i - d2 * HH;     // w2t[d][h] = W2[h][d]
  w2t[i] = (_Float16)W2[h2 * DD + d2];
}

__global__ __launch_bounds__(512)
void joiner_main(const float* __restrict__ src, const float* __restrict__ tgt,
                 const _Float16* __restrict__ w1t, const float* __restrict__ b1,
                 const _Float16* __restrict__ w2t, const float* __restrict__ b2,
                 float* __restrict__ out) {
  __shared__ _Float16 sA[ROWS * SA_STRIDE];  // 32x512 f16 A tile  (~33.3 KB)
  __shared__ _Float16 sH[ROWS * SH_STRIDE];  // 32x256 f16 h chunk (~16.9 KB)

  const int t     = blockIdx.x >> 1;
  const int uhalf = blockIdx.x & 1;
  const int b     = blockIdx.y;
  const int u0    = uhalf * ROWS;

  const int tid   = threadIdx.x;
  const int wave  = tid >> 5;      // 0..15
  const int lane  = tid & 31;
  const int lmod  = lane & 15;     // column / row-in-tile index
  const int lhalf = lane >> 4;     // 0 or 1 (selects K sub-block per WMMA layout)

  // ---- Stage A = src[b,t,:] (broadcast) + tgt[b,u0+u,:] as f16 in LDS ----
  const float* srcRow = src + ((size_t)(b * TT + t)) * DD;
  const float* tgtMat = tgt + ((size_t)(b * UU + u0)) * DD;
  for (int i = tid; i < ROWS * DD; i += 512) {
    int u = i >> 9;            // /512
    int d = i & (DD - 1);
    sA[u * SA_STRIDE + d] = (_Float16)(srcRow[d] + tgtMat[(size_t)u * DD + d]);
  }
  __syncthreads();

  // Output accumulators: wave owns out cols [wave*32, wave*32+32), all 32 rows.
  // 2 M-tiles x 2 N-tiles of 16x16 f32 -> 4 x v8f = 32 VGPRs.
  v8f acc[2][2];
  {
    v8f z = {};
    acc[0][0] = z; acc[0][1] = z; acc[1][0] = z; acc[1][1] = z;
  }

  const int n1 = wave * 16 + lmod;  // within-chunk H column for GEMM1 (16 waves x 16 = 256)

  #pragma unroll 1
  for (int hb = 0; hb < HH; hb += HCHUNK) {
    // ---------------- GEMM1: h[:, hb+n1 block] = A(32x512) x W1 ----------------
    v8f hacc[2];
    { v8f z = {}; hacc[0] = z; hacc[1] = z; }

    const _Float16* w1row = w1t + (size_t)(hb + n1) * DD + lhalf * 16;
    #pragma unroll 4
    for (int k0 = 0; k0 < DD; k0 += 32) {
      v16h bf = *(const v16h*)(w1row + k0);   // B frag: col n1, K contiguous 32B
      #pragma unroll
      for (int mt = 0; mt < 2; ++mt) {
        union { v16h v; v8h h[2]; } a;        // A frag, K contiguous per lane half (same as B)
        const _Float16* ar = &sA[(mt * 16 + lmod) * SA_STRIDE + k0 + lhalf * 16];
        a.h[0] = *(const v8h*)(ar);           // K = k0 + {0..7}  (+16 if lane>=16)
        a.h[1] = *(const v8h*)(ar + 8);       // K = k0 + 8 + {0..7} (+16 ...)
        hacc[mt] = __builtin_amdgcn_wmma_f32_16x16x32_f16(
            false, a.v, false, bf, (short)0, hacc[mt], false, false);
      }
    }

    // Bias + tanh (branch-free native v_tanh_f32), spill chunk to LDS as f16
    {
      float bias1 = b1[hb + n1];
      #pragma unroll
      for (int mt = 0; mt < 2; ++mt) {
        #pragma unroll
        for (int v = 0; v < 8; ++v) {
          int row = mt * 16 + v + lhalf * 8;
          float x = hacc[mt][v] + bias1;
          sH[row * SH_STRIDE + n1] = (_Float16)fast_tanh(x);
        }
      }
    }
    __syncthreads();

    // ---------------- GEMM2 partial: acc += tanh_chunk(32x256) x W2 ----------------
    #pragma unroll 4
    for (int k0 = 0; k0 < HCHUNK; k0 += 32) {
      union { v16h v; v8h h[2]; } af[2];
      #pragma unroll
      for (int mt = 0; mt < 2; ++mt) {
        const _Float16* ar = &sH[(mt * 16 + lmod) * SH_STRIDE + k0 + lhalf * 16];
        af[mt].h[0] = *(const v8h*)(ar);
        af[mt].h[1] = *(const v8h*)(ar + 8);
      }
      #pragma unroll
      for (int nt = 0; nt < 2; ++nt) {
        int n = wave * 32 + nt * 16 + lmod;   // output D column
        v16h bf2 = *(const v16h*)(w2t + (size_t)n * HH + hb + k0 + lhalf * 16);
        #pragma unroll
        for (int mt = 0; mt < 2; ++mt)
          acc[mt][nt] = __builtin_amdgcn_wmma_f32_16x16x32_f16(
              false, af[mt].v, false, bf2, (short)0, acc[mt][nt], false, false);
      }
    }
    __syncthreads();  // protect sH before next chunk overwrites it
  }

  // ---------------- Epilogue: + b2, store fp32 ----------------
  float* outBase = out + (((size_t)(b * TT + t)) * UU + u0) * DD;
  #pragma unroll
  for (int nt = 0; nt < 2; ++nt) {
    int n = wave * 32 + nt * 16 + lmod;
    float bias2 = b2[n];
    #pragma unroll
    for (int mt = 0; mt < 2; ++mt) {
      #pragma unroll
      for (int v = 0; v < 8; ++v) {
        int u = mt * 16 + v + lhalf * 8;
        outBase[(size_t)u * DD + n] = acc[mt][nt][v] + bias2;
      }
    }
  }
}

extern "C" void kernel_launch(void* const* d_in, const int* in_sizes, int n_in,
                              void* d_out, int out_size, void* d_ws, size_t ws_size,
                              hipStream_t stream) {
  (void)in_sizes; (void)n_in;
  const float* src  = (const float*)d_in[0];
  const int*   slen = (const int*)d_in[1];
  const float* tgt  = (const float*)d_in[2];
  const int*   tlen = (const int*)d_in[3];
  const float* W1   = (const float*)d_in[4];
  const float* b1   = (const float*)d_in[5];
  const float* W2   = (const float*)d_in[6];
  const float* b2   = (const float*)d_in[7];
  float* outp = (float*)d_out;

  // Workspace: f16 W1^T (H*D) followed by f16 W2^T (D*H) = 2 MB total.
  size_t need = 2 * (size_t)DD * HH * sizeof(_Float16);
  if (ws_size < need) return;
  _Float16* w1t = (_Float16*)d_ws;
  _Float16* w2t = w1t + (size_t)DD * HH;

  joiner_prep<<<(DD * HH + 511) / 512, 512, 0, stream>>>(W1, W2, w1t, w2t);

  dim3 grid(TT * 2, BB);  // (t, u-half) x batch
  joiner_main<<<grid, 512, 0, stream>>>(src, tgt, w1t, b1, w2t, b2, outp);

  // Tuple outputs: out (B*T*U*D f32) ++ source_lengths ++ target_lengths (raw i32 bits)
  size_t mainElems = (size_t)BB * TT * UU * DD;
  if ((size_t)out_size >= mainElems + 2 * BB) {
    hipMemcpyAsync((char*)d_out + mainElems * sizeof(float), slen,
                   BB * sizeof(int), hipMemcpyDeviceToDevice, stream);
    hipMemcpyAsync((char*)d_out + mainElems * sizeof(float) + BB * sizeof(int), tlen,
                   BB * sizeof(int), hipMemcpyDeviceToDevice, stream);
  }
}
